// MambaBlock_45896020525187
// MI455X (gfx1250) — matrix operand order repaired and checked by reading, not compiled, not run on hardware
//
#include <hip/hip_runtime.h>
#include <hip/hip_bf16.h>

// ---------------------------------------------------------------------------
// Mamba block for MI455X (gfx1250, wave32, WMMA + async-to-LDS pipeline).
// Shapes: B=4, L=1024, DIM=1024, D_INNER=2048, D_STATE=16, D_CONV=4, DT_RANK=64
// ---------------------------------------------------------------------------

#define T_TOKENS 4096
#define DIM_     1024
#define DINNER   2048
#define DSTATE   16
#define DCONV    4
#define DTRANK   64
#define LSEQ     1024
#define BATCH    4

typedef __attribute__((ext_vector_type(16))) __bf16 v16bf;
typedef __attribute__((ext_vector_type(8)))  float  v8f;

union FragBF { uint4 q[2]; v16bf v; };

__device__ __forceinline__ unsigned short f2bf(float f) {
    union { float f; unsigned int u; } c; c.f = f;
    unsigned int u = c.u;
    unsigned int r = u + 0x7FFFu + ((u >> 16) & 1u);   // RNE
    return (unsigned short)(r >> 16);
}

__device__ __forceinline__ float silu_f(float x) {
    return x / (1.0f + __expf(-x));
}

// CDNA5 async global->LDS copy, 16B per lane, tracked by ASYNCcnt.
__device__ __forceinline__ void async_cp16(void* lds, const void* gptr) {
    asm volatile("global_load_async_to_lds_b128 %0, %1, off"
                 :: "v"((unsigned)(size_t)lds), "v"(gptr)
                 : "memory");
}
__device__ __forceinline__ void wait_async3() {
    asm volatile("s_wait_asynccnt 0x3" ::: "memory");
}
__device__ __forceinline__ void wait_async0() {
    asm volatile("s_wait_asynccnt 0x0" ::: "memory");
}

// ---------------------------------------------------------------------------
// fp32 -> bf16 cast (weights)
// ---------------------------------------------------------------------------
__global__ __launch_bounds__(256) void cast_bf16_kernel(const float* __restrict__ src,
                                                        unsigned short* __restrict__ dst,
                                                        int n) {
    for (int i = blockIdx.x * 256 + threadIdx.x; i < n; i += gridDim.x * 256)
        dst[i] = f2bf(src[i]);
}

// ---------------------------------------------------------------------------
// RMSNorm over DIM=1024 per token, output bf16
// ---------------------------------------------------------------------------
__global__ __launch_bounds__(256) void rmsnorm_cast_kernel(const float* __restrict__ x,
                                                           const float* __restrict__ w,
                                                           unsigned short* __restrict__ h) {
    int token = blockIdx.x;
    int tid   = threadIdx.x;
    const float* xr = x + (size_t)token * DIM_;
    float v[4]; float ss = 0.f;
#pragma unroll
    for (int i = 0; i < 4; i++) { v[i] = xr[tid + 256 * i]; ss += v[i] * v[i]; }
#pragma unroll
    for (int off = 16; off; off >>= 1) ss += __shfl_xor(ss, off, 32);
    __shared__ float wsum[8];
    if ((tid & 31) == 0) wsum[tid >> 5] = ss;
    __syncthreads();
    float tot = 0.f;
#pragma unroll
    for (int i = 0; i < 8; i++) tot += wsum[i];
    float norm = rsqrtf(tot * (1.0f / DIM_) + 1e-6f);
#pragma unroll
    for (int i = 0; i < 4; i++) {
        int c = tid + 256 * i;
        h[(size_t)token * DIM_ + c] = f2bf(v[i] * norm * w[c]);
    }
}

// ---------------------------------------------------------------------------
// LDS-staged WMMA GEMM: C[M,N] = A[M,K] * W[N,K]^T (+ optional residual)
// Block = 256 threads = 8 waves arranged 2(M) x 4(N).
// Block tile 64(M) x 128(N) x 32(K); per-wave tile 32x32 (4 accumulators).
// Tiles staged via global_load_async_to_lds_b128, double buffered.  Peeled
// software pipeline: main loop unconditionally issues tile kt+1 then waits
// asynccnt<=3 (tile kt done, next still in flight); last tile peeled out.
// ---------------------------------------------------------------------------
#define BM 64
#define BN 128
#define BK 32

__global__ __launch_bounds__(256) void wmma_gemm_lds_kernel(const unsigned short* __restrict__ A,
                                                            const unsigned short* __restrict__ W,
                                                            float* __restrict__ C,
                                                            const float* __restrict__ resid,
                                                            int M, int N, int K) {
    __shared__ alignas(16) unsigned short As[2][BM][BK];   //  8 KB
    __shared__ alignas(16) unsigned short Ws[2][BN][BK];   // 16 KB

    const int tid  = threadIdx.x;
    const int lane = tid & 31;
    const int half = lane >> 4;
    const int r    = lane & 15;
    const int wave = tid >> 5;
    const int wm   = wave >> 2;       // 0..1
    const int wn   = wave & 3;        // 0..3

    const int tilesN = N / BN;
    const int bm = blockIdx.x / tilesN;
    const int bn = blockIdx.x % tilesN;
    const int m0 = bm * BM;
    const int n0 = bn * BN;

    // copy-index decomposition: 16B segments, 4 per 64B row
    const int arow = tid >> 2, aseg = tid & 3;
    const int KT = K / BK;

    v8f acc[2][2];
    const v8f zero = {0.f, 0.f, 0.f, 0.f, 0.f, 0.f, 0.f, 0.f};
#pragma unroll
    for (int mi = 0; mi < 2; mi++)
#pragma unroll
        for (int ni = 0; ni < 2; ni++) acc[mi][ni] = zero;

    // stage one k-tile (3 async copies per thread: 1 A seg + 2 W segs)
    auto stage = [&](int buf, int k0) {
        async_cp16(&As[buf][arow][aseg * 8], A + (size_t)(m0 + arow) * K + k0 + aseg * 8);
#pragma unroll
        for (int t = 0; t < 2; t++) {
            int s = tid + t * 256;
            int row = s >> 2, seg = s & 3;
            async_cp16(&Ws[buf][row][seg * 8], W + (size_t)(n0 + row) * K + k0 + seg * 8);
        }
    };

    // ds_load fragments from buffer `buf` and issue 4 WMMAs
    auto compute = [&](int buf) {
        FragBF fa[2];
#pragma unroll
        for (int mi = 0; mi < 2; mi++) {
            const unsigned short* ar = &As[buf][wm * 32 + mi * 16 + r][0];
            fa[mi].q[0] = *(const uint4*)(ar + 8 * half);
            fa[mi].q[1] = *(const uint4*)(ar + 16 + 8 * half);
        }
#pragma unroll
        for (int ni = 0; ni < 2; ni++) {
            const unsigned short* wr = &Ws[buf][wn * 32 + ni * 16 + r][0];
            FragBF fb;
            fb.q[0] = *(const uint4*)(wr + 8 * half);
            fb.q[1] = *(const uint4*)(wr + 16 + 8 * half);
#pragma unroll
            for (int mi = 0; mi < 2; mi++)
                acc[mi][ni] = __builtin_amdgcn_wmma_f32_16x16x32_bf16(
                    false, fa[mi].v, false, fb.v, (short)0, acc[mi][ni], false, false);
        }
    };

    // prologue: tile 0 in flight
    stage(0, 0);

    // main pipeline (KT-1 iterations, branch-free body)
    for (int kt = 0; kt < KT - 1; kt++) {
        const int buf = kt & 1;
        stage(buf ^ 1, (kt + 1) * BK);   // next tile in flight
        wait_async3();                   // current tile complete (in-order)
        __syncthreads();                 // copies visible to all threads
        compute(buf);
        __syncthreads();                 // protect buffer reuse at kt+2
    }

    // epilogue: last tile
    wait_async0();
    __syncthreads();
    compute((KT - 1) & 1);

#pragma unroll
    for (int mi = 0; mi < 2; mi++)
#pragma unroll
        for (int ni = 0; ni < 2; ni++) {
            const int row0 = m0 + wm * 32 + mi * 16 + 8 * half;
            const int col  = n0 + wn * 32 + ni * 16 + r;
            float* crow = C + (size_t)row0 * N + col;
#pragma unroll
            for (int rr = 0; rr < 8; rr++) {
                float v = acc[mi][ni][rr];
                if (resid) v += resid[(size_t)(row0 + rr) * N + col];
                crow[(size_t)rr * N] = v;
            }
        }
}

// ---------------------------------------------------------------------------
// Direct-load WMMA GEMM for the skinny x_proj (N=96): per-wave 16 x (16*NT)
// ---------------------------------------------------------------------------
template <int NT>
__global__ __launch_bounds__(256) void wmma_gemm_kernel(const unsigned short* __restrict__ A,
                                                        const unsigned short* __restrict__ W,
                                                        float* __restrict__ C,
                                                        int M, int N, int K) {
    const int lane  = threadIdx.x & 31;
    const int half  = lane >> 4;
    const int r     = lane & 15;
    const int wave  = blockIdx.x * 8 + (threadIdx.x >> 5);
    const int tilesN = N / (16 * NT);
    const int m0 = (wave / tilesN) * 16;
    const int n0 = (wave % tilesN) * 16 * NT;

    v8f acc[NT];
    const v8f zero = {0.f, 0.f, 0.f, 0.f, 0.f, 0.f, 0.f, 0.f};
#pragma unroll
    for (int j = 0; j < NT; j++) acc[j] = zero;

    const unsigned short* arow = A + (size_t)(m0 + r) * K;

    for (int k0 = 0; k0 < K; k0 += 32) {
        FragBF fa;
        fa.q[0] = *(const uint4*)(arow + k0 + 8 * half);
        fa.q[1] = *(const uint4*)(arow + k0 + 16 + 8 * half);
        if (k0 + 32 < K)
            __builtin_prefetch((const void*)(arow + k0 + 32), 0, 1);
#pragma unroll
        for (int j = 0; j < NT; j++) {
            const unsigned short* wrow = W + (size_t)(n0 + j * 16 + r) * K;
            FragBF fb;
            fb.q[0] = *(const uint4*)(wrow + k0 + 8 * half);
            fb.q[1] = *(const uint4*)(wrow + k0 + 16 + 8 * half);
            acc[j] = __builtin_amdgcn_wmma_f32_16x16x32_bf16(
                false, fa.v, false, fb.v, (short)0, acc[j], false, false);
        }
    }

#pragma unroll
    for (int j = 0; j < NT; j++) {
        float* crow = C + (size_t)(m0 + 8 * half) * N + n0 + j * 16 + r;
#pragma unroll
        for (int rr = 0; rr < 8; rr++)
            crow[(size_t)rr * N] = acc[j][rr];
    }
}

// ---------------------------------------------------------------------------
// Depthwise causal conv (width 4) + bias + SiLU; writes f32 and bf16 copies.
// ---------------------------------------------------------------------------
__global__ __launch_bounds__(256) void conv_silu_kernel(const float* __restrict__ xz,
                                                        const float* __restrict__ cw,
                                                        const float* __restrict__ cb,
                                                        float* __restrict__ xc,
                                                        unsigned short* __restrict__ xcb) {
    const int total = T_TOKENS * DINNER;
    for (int idx = blockIdx.x * 256 + threadIdx.x; idx < total; idx += gridDim.x * 256) {
        int d     = idx & (DINNER - 1);
        int token = idx >> 11;
        int l     = token & (LSEQ - 1);
        float acc = cb[d];
#pragma unroll
        for (int j = 0; j < DCONV; j++) {
            int lt = l - (DCONV - 1) + j;
            if (lt >= 0)
                acc = fmaf(cw[d * DCONV + j],
                           xz[(size_t)(token - (DCONV - 1) + j) * (2 * DINNER) + d], acc);
        }
        float s = silu_f(acc);
        xc[idx]  = s;
        xcb[idx] = f2bf(s);
    }
}

// ---------------------------------------------------------------------------
// Slice dt low-rank (first 64 cols of x_dbl) -> bf16
// ---------------------------------------------------------------------------
__global__ __launch_bounds__(256) void dtlow_cast_kernel(const float* __restrict__ xdbl,
                                                         unsigned short* __restrict__ dtlow) {
    const int total = T_TOKENS * DTRANK;
    for (int idx = blockIdx.x * 256 + threadIdx.x; idx < total; idx += gridDim.x * 256) {
        int token = idx >> 6;
        int c     = idx & (DTRANK - 1);
        dtlow[idx] = f2bf(xdbl[(size_t)token * (DTRANK + 2 * DSTATE) + c]);
    }
}

// ---------------------------------------------------------------------------
// dt = softplus(dt_raw + bias), in place
// ---------------------------------------------------------------------------
__global__ __launch_bounds__(256) void softplus_kernel(float* __restrict__ dt,
                                                       const float* __restrict__ bias) {
    const int total = T_TOKENS * DINNER;
    for (int idx = blockIdx.x * 256 + threadIdx.x; idx < total; idx += gridDim.x * 256) {
        int d   = idx & (DINNER - 1);
        float v = dt[idx] + bias[d];
        dt[idx] = (v > 20.f) ? v : log1pf(__expf(v));
    }
}

// ---------------------------------------------------------------------------
// Selective scan: grid (DINNER/256, BATCH); one channel per lane, 16-element
// register state; per-step B/C in double-buffered LDS (one barrier per step).
// ---------------------------------------------------------------------------
__global__ __launch_bounds__(256) void scan_kernel(const float* __restrict__ dt,
                                                   const float* __restrict__ xc,
                                                   const float* __restrict__ xdbl,
                                                   const float* __restrict__ A_log,
                                                   const float* __restrict__ Dskip,
                                                   float* __restrict__ ybuf) {
    const int b = blockIdx.y;
    const int d = blockIdx.x * 256 + threadIdx.x;

    float Ad[DSTATE];
#pragma unroll
    for (int n = 0; n < DSTATE; n++) Ad[n] = -__expf(A_log[(size_t)d * DSTATE + n]);
    const float Dd = Dskip[d];

    float s[DSTATE];
#pragma unroll
    for (int n = 0; n < DSTATE; n++) s[n] = 0.f;

    __shared__ float bc[2][2 * DSTATE];

    for (int t = 0; t < LSEQ; t++) {
        const int tok = b * LSEQ + t;
        if (threadIdx.x < 2 * DSTATE)
            bc[t & 1][threadIdx.x] =
                xdbl[(size_t)tok * (DTRANK + 2 * DSTATE) + DTRANK + threadIdx.x];
        __syncthreads();
        const float* Bv = bc[t & 1];
        const float* Cv = bc[t & 1] + DSTATE;
        const float dtv = dt[(size_t)tok * DINNER + d];
        const float u   = xc[(size_t)tok * DINNER + d];
        const float du  = dtv * u;
        float y = 0.f;
#pragma unroll
        for (int n = 0; n < DSTATE; n++) {
            s[n] = fmaf(s[n], __expf(dtv * Ad[n]), du * Bv[n]);
            y    = fmaf(s[n], Cv[n], y);
        }
        ybuf[(size_t)tok * DINNER + d] = fmaf(Dd, u, y);
    }
}

// ---------------------------------------------------------------------------
// Gate: y *= silu(z); output bf16 for out_proj GEMM
// ---------------------------------------------------------------------------
__global__ __launch_bounds__(256) void gate_kernel(const float* __restrict__ ybuf,
                                                   const float* __restrict__ xz,
                                                   unsigned short* __restrict__ ybb) {
    const int total = T_TOKENS * DINNER;
    for (int idx = blockIdx.x * 256 + threadIdx.x; idx < total; idx += gridDim.x * 256) {
        int d     = idx & (DINNER - 1);
        int token = idx >> 11;
        float z = xz[(size_t)token * (2 * DINNER) + DINNER + d];
        ybb[idx] = f2bf(ybuf[idx] * silu_f(z));
    }
}

// ---------------------------------------------------------------------------
// Host orchestration
// ---------------------------------------------------------------------------
extern "C" void kernel_launch(void* const* d_in, const int* in_sizes, int n_in,
                              void* d_out, int out_size, void* d_ws, size_t ws_size,
                              hipStream_t stream) {
    (void)in_sizes; (void)n_in; (void)out_size; (void)ws_size;

    const float* x        = (const float*)d_in[0];
    const float* rms_w    = (const float*)d_in[1];
    const float* in_w     = (const float*)d_in[2];   // (4096, 1024)
    const float* conv_w   = (const float*)d_in[3];   // (2048, 1, 4)
    const float* conv_b   = (const float*)d_in[4];
    const float* xproj_w  = (const float*)d_in[5];   // (96, 2048)
    const float* dtproj_w = (const float*)d_in[6];   // (2048, 64)
    const float* dtproj_b = (const float*)d_in[7];
    const float* A_log    = (const float*)d_in[8];   // (2048, 16)
    const float* Dskip    = (const float*)d_in[9];
    const float* out_w    = (const float*)d_in[10];  // (1024, 2048)
    float* out = (float*)d_out;

    char* p = (char*)d_ws;
    auto carve = [&](size_t bytes) -> void* {
        void* r = (void*)p;
        p += (bytes + 255) & ~(size_t)255;
        return r;
    };
    unsigned short* hb     = (unsigned short*)carve((size_t)T_TOKENS * DIM_ * 2);
    unsigned short* winb   = (unsigned short*)carve((size_t)(2 * DINNER) * DIM_ * 2);
    unsigned short* xpwb   = (unsigned short*)carve((size_t)(DTRANK + 2 * DSTATE) * DINNER * 2);
    unsigned short* dtpwb  = (unsigned short*)carve((size_t)DINNER * DTRANK * 2);
    unsigned short* outpwb = (unsigned short*)carve((size_t)DIM_ * DINNER * 2);
    float*          xz     = (float*)carve((size_t)T_TOKENS * 2 * DINNER * 4);
    float*          xc     = (float*)carve((size_t)T_TOKENS * DINNER * 4);
    unsigned short* xcb    = (unsigned short*)carve((size_t)T_TOKENS * DINNER * 2);
    float*          xdbl   = (float*)carve((size_t)T_TOKENS * (DTRANK + 2 * DSTATE) * 4);
    unsigned short* dtlowb = (unsigned short*)carve((size_t)T_TOKENS * DTRANK * 2);
    float*          dtbuf  = (float*)carve((size_t)T_TOKENS * DINNER * 4);
    float*          ybuf   = (float*)carve((size_t)T_TOKENS * DINNER * 4);
    unsigned short* ybb    = (unsigned short*)carve((size_t)T_TOKENS * DINNER * 2);

    auto eb = [](int n) { return (n + 255) / 256; };

    // 1) weight casts fp32 -> bf16
    cast_bf16_kernel<<<eb(2 * DINNER * DIM_), 256, 0, stream>>>(in_w, winb, 2 * DINNER * DIM_);
    cast_bf16_kernel<<<eb((DTRANK + 2 * DSTATE) * DINNER), 256, 0, stream>>>(
        xproj_w, xpwb, (DTRANK + 2 * DSTATE) * DINNER);
    cast_bf16_kernel<<<eb(DINNER * DTRANK), 256, 0, stream>>>(dtproj_w, dtpwb, DINNER * DTRANK);
    cast_bf16_kernel<<<eb(DIM_ * DINNER), 256, 0, stream>>>(out_w, outpwb, DIM_ * DINNER);

    // 2) RMSNorm -> bf16
    rmsnorm_cast_kernel<<<T_TOKENS, 256, 0, stream>>>(x, rms_w, hb);

    // 3) in_proj: (4096,1024) x (4096,1024)^T -> (4096,4096)   [LDS/async WMMA]
    {
        int M = T_TOKENS, N = 2 * DINNER, K = DIM_;
        int blocks = (M / BM) * (N / BN);
        wmma_gemm_lds_kernel<<<blocks, 256, 0, stream>>>(hb, winb, xz, nullptr, M, N, K);
    }

    // 4) causal depthwise conv + SiLU
    conv_silu_kernel<<<eb(T_TOKENS * DINNER), 256, 0, stream>>>(xz, conv_w, conv_b, xc, xcb);

    // 5) x_proj: (4096,2048) x (96,2048)^T -> (4096,96)   [direct WMMA, NT=2]
    {
        int M = T_TOKENS, N = DTRANK + 2 * DSTATE, K = DINNER;
        int waves = (M / 16) * (N / 32);
        wmma_gemm_kernel<2><<<waves / 8, 256, 0, stream>>>(xcb, xpwb, xdbl, M, N, K);
    }

    // 6) dt low-rank slice -> bf16
    dtlow_cast_kernel<<<eb(T_TOKENS * DTRANK), 256, 0, stream>>>(xdbl, dtlowb);

    // 7) dt_proj: (4096,64) x (2048,64)^T -> (4096,2048)   [LDS/async WMMA]
    {
        int M = T_TOKENS, N = DINNER, K = DTRANK;
        int blocks = (M / BM) * (N / BN);
        wmma_gemm_lds_kernel<<<blocks, 256, 0, stream>>>(dtlowb, dtpwb, dtbuf, nullptr, M, N, K);
    }

    // 8) softplus(dt + b)
    softplus_kernel<<<eb(T_TOKENS * DINNER), 256, 0, stream>>>(dtbuf, dtproj_b);

    // 9) selective scan (+ Dskip*u fused)
    {
        dim3 grid(DINNER / 256, BATCH);
        scan_kernel<<<grid, 256, 0, stream>>>(dtbuf, xc, xdbl, A_log, Dskip, ybuf);
    }

    // 10) gate with silu(z) -> bf16
    gate_kernel<<<eb(T_TOKENS * DINNER), 256, 0, stream>>>(ybuf, xz, ybb);

    // 11) out_proj + fused residual: out = x + ybb x out_w^T   [LDS/async WMMA]
    {
        int M = T_TOKENS, N = DIM_, K = DINNER;
        int blocks = (M / BM) * (N / BN);
        wmma_gemm_lds_kernel<<<blocks, 256, 0, stream>>>(ybb, outpwb, out, x, M, N, K);
    }
}
